// PointTransformer_37091337568772
// MI455X (gfx1250) — matrix operand order repaired and checked by reading, not compiled, not run on hardware
//
#include <hip/hip_runtime.h>
#include <hip/hip_bf16.h>

// ---------------------------------------------------------------------------
// Types for CDNA5 WMMA (wave32).
// ---------------------------------------------------------------------------
typedef __attribute__((ext_vector_type(16))) __bf16 v16bf;
typedef __attribute__((ext_vector_type(8)))  float  v8f;

union Frag { v16bf v; uint4 q[2]; };

__device__ __forceinline__ unsigned bfbits(float x) {
  union { float f; unsigned u; } c; c.f = x;
  return ((c.u + 0x7FFFu + ((c.u >> 16) & 1u)) >> 16) & 0xFFFFu;  // RNE
}
__device__ __forceinline__ unsigned packbf(float lo, float hi) {
  return bfbits(lo) | (bfbits(hi) << 16);
}

// ---------------------------------------------------------------------------
// WMMA GEMM:  C[M,N] = act(A[M,K] @ W[K,N] + bias)
// Block = 128 threads = 4 waves. Block tile 128(M) x 64(N); each wave owns a
// 32x64 strip: 2 A-fragments x 4 B-subtiles = 8 accumulators -> 8 v_wmma per
// K-step per wave.
//
// LDS holds bf16 K-pairs packed into dwords, in fragment dword order
// (ISA 7.12.2): fragment dword d of A covers K = {2d,2d+1} (d<4) /
// {16+2(d-4),...} (+8 per lane-half), so a lane's 16-element fragment is two
// contiguous 16B chunks -> 2x ds_load_b128 per fragment.
// ---------------------------------------------------------------------------
#define GBM 128
#define GBN 64
#define GBK 32

template <int ACT>   // 0 = none, 1 = relu
__global__ __launch_bounds__(128) void gemm_wmma_kernel(
    const float* __restrict__ A, const float* __restrict__ W,
    const float* __restrict__ bias, float* __restrict__ C,
    int M, int N, int K) {
  __shared__ unsigned sAp[GBM][GBK / 2];   // [row][k2]  8 KB
  __shared__ unsigned sBp[GBN][GBK / 2];   // [col][k2]  4 KB
  const int tid  = threadIdx.x;
  const int wave = tid >> 5;
  const int lane = tid & 31;
  const int l15  = lane & 15;
  const int half = lane >> 4;
  const int m0 = blockIdx.x * GBM;
  const int n0 = blockIdx.y * GBN;

  const bool interiorM = (m0 + GBM <= M);
  const bool k4 = ((K & 3) == 0);

  v8f acc[2][4] = {};

  for (int k0 = 0; k0 < K; k0 += GBK) {
    const bool fast = interiorM && k4 && (k0 + GBK <= K);
    if (fast) {
      // ---- A tile: 128 rows x 8 float4 each = 1024 float4, 8 per thread
      #pragma unroll
      for (int it = 0; it < 8; ++it) {
        int p = tid + it * 128;          // 0..1023
        int r = p >> 3, c4 = p & 7;      // row, float4-chunk
        const float4 f = *(const float4*)(A + (size_t)(m0 + r) * K + k0 + 4 * c4);
        sAp[r][2 * c4]     = packbf(f.x, f.y);
        sAp[r][2 * c4 + 1] = packbf(f.z, f.w);
      }
      // ---- B tile: 16 k2 x 16 col4 groups = 256, 2 per thread
      #pragma unroll
      for (int it = 0; it < 2; ++it) {
        int p = tid + it * 128;          // 0..255
        int k2 = p >> 4, n4 = p & 15;    // k-pair, 4-col group
        const float* w0 = W + (size_t)(k0 + 2 * k2) * N + n0 + 4 * n4;
        const float4 e = *(const float4*)w0;        // K = 2*k2
        const float4 o = *(const float4*)(w0 + N);  // K = 2*k2+1
        sBp[4 * n4 + 0][k2] = packbf(e.x, o.x);
        sBp[4 * n4 + 1][k2] = packbf(e.y, o.y);
        sBp[4 * n4 + 2][k2] = packbf(e.z, o.z);
        sBp[4 * n4 + 3][k2] = packbf(e.w, o.w);
      }
    } else {
      // ---- branchless guarded staging (clamped address, select after load)
      #pragma unroll
      for (int i = tid; i < GBM * (GBK / 2); i += 128) {
        int r = i >> 4, c2 = i & 15;
        int gr = m0 + r, gc0 = k0 + 2 * c2, gc1 = gc0 + 1;
        int grc = gr < M ? gr : M - 1;
        float u0 = A[(size_t)grc * K + (gc0 < K ? gc0 : K - 1)];
        float u1 = A[(size_t)grc * K + (gc1 < K ? gc1 : K - 1)];
        float v0 = (gr < M && gc0 < K) ? u0 : 0.0f;
        float v1 = (gr < M && gc1 < K) ? u1 : 0.0f;
        sAp[r][c2] = packbf(v0, v1);
      }
      #pragma unroll
      for (int i = tid; i < GBN * (GBK / 2); i += 128) {
        int n = i >> 4, c2 = i & 15;
        int gn = n0 + n, gk0 = k0 + 2 * c2, gk1 = gk0 + 1;
        int gnc = gn < N ? gn : N - 1;
        float u0 = W[(size_t)(gk0 < K ? gk0 : K - 1) * N + gnc];
        float u1 = W[(size_t)(gk1 < K ? gk1 : K - 1) * N + gnc];
        float v0 = (gn < N && gk0 < K) ? u0 : 0.0f;
        float v1 = (gn < N && gk1 < K) ? u1 : 0.0f;
        sBp[n][c2] = packbf(v0, v1);
      }
    }
    __syncthreads();

    // ---- fragments: 2x b128 each; 8 WMMAs reusing 2 A-fragments
    Frag a[2];
    #pragma unroll
    for (int mi = 0; mi < 2; ++mi) {
      const int arow = wave * 32 + mi * 16 + l15;
      a[mi].q[0] = *(const uint4*)&sAp[arow][half * 4];
      a[mi].q[1] = *(const uint4*)&sAp[arow][8 + half * 4];
    }
    #pragma unroll
    for (int nt = 0; nt < 4; ++nt) {
      Frag b;
      b.q[0] = *(const uint4*)&sBp[nt * 16 + l15][half * 4];
      b.q[1] = *(const uint4*)&sBp[nt * 16 + l15][8 + half * 4];
      #pragma unroll
      for (int mi = 0; mi < 2; ++mi) {
        acc[mi][nt] = __builtin_amdgcn_wmma_f32_16x16x32_bf16(
            false, a[mi].v, false, b.v, (short)0, acc[mi][nt], false, false);
      }
    }
    __syncthreads();
  }

  // ---- epilogue: C VGPR r -> row (r + 8*half), col = lane&15 within subtile
  #pragma unroll
  for (int nt = 0; nt < 4; ++nt) {
    const int col = n0 + nt * 16 + l15;
    if (col < N) {
      const float bv = bias ? bias[col] : 0.0f;
      #pragma unroll
      for (int mi = 0; mi < 2; ++mi) {
        #pragma unroll
        for (int r = 0; r < 8; ++r) {
          int row = m0 + wave * 32 + mi * 16 + r + half * 8;
          if (row < M) {
            float v = acc[mi][nt][r] + bv;
            if (ACT == 1) v = fmaxf(v, 0.0f);
            C[(size_t)row * N + col] = v;
          }
        }
      }
    }
  }
}

// ---------------------------------------------------------------------------
// Brute-force KNN: one block (128 threads) per query point.
// Per-thread register top-KK insertion list, then LDS tree-merge (KK rounds).
// ---------------------------------------------------------------------------
template <int KK>
__global__ __launch_bounds__(128) void knn_kernel(
    const float* __restrict__ ref, const float* __restrict__ query,
    int Nref, int Nq, int* __restrict__ out_idx, float* __restrict__ out_d) {
  const int q = blockIdx.x;
  const int b = blockIdx.y;
  const int tid = threadIdx.x;
  const float* refb = ref + (size_t)b * Nref * 3;
  const float* qp   = query + ((size_t)b * Nq + q) * 3;
  const float qx = qp[0], qy = qp[1], qz = qp[2];

  float bd[KK]; int bi[KK];
  #pragma unroll
  for (int i = 0; i < KK; ++i) { bd[i] = 3.4e38f; bi[i] = 0; }

  for (int r = tid; r < Nref; r += 128) {
    float dx = qx - refb[r * 3 + 0];
    float dy = qy - refb[r * 3 + 1];
    float dz = qz - refb[r * 3 + 2];
    float d2 = dx * dx + dy * dy + dz * dz;
    if (d2 < bd[KK - 1]) {
      bd[KK - 1] = d2; bi[KK - 1] = r;
      #pragma unroll
      for (int i = KK - 2; i >= 0; --i) {
        if (bd[i + 1] < bd[i]) {
          float td = bd[i]; bd[i] = bd[i + 1]; bd[i + 1] = td;
          int   ti = bi[i]; bi[i] = bi[i + 1]; bi[i + 1] = ti;
        }
      }
    }
  }

  __shared__ float sd[128 * KK];
  __shared__ int   si[128 * KK];
  __shared__ int   shead[128];
  __shared__ float rv[128];
  __shared__ int   rt[128];
  #pragma unroll
  for (int i = 0; i < KK; ++i) { sd[tid * KK + i] = bd[i]; si[tid * KK + i] = bi[i]; }
  shead[tid] = 0;
  __syncthreads();

  const size_t obase = ((size_t)b * Nq + q) * KK;
  for (int sel = 0; sel < KK; ++sel) {
    int h = shead[tid];
    rv[tid] = (h < KK) ? sd[tid * KK + h] : 3.4e38f;
    rt[tid] = tid;
    __syncthreads();
    for (int s = 64; s > 0; s >>= 1) {
      if (tid < s && rv[tid + s] < rv[tid]) { rv[tid] = rv[tid + s]; rt[tid] = rt[tid + s]; }
      __syncthreads();
    }
    if (tid == 0) {
      int bt = rt[0];
      out_idx[obase + sel] = si[bt * KK + shead[bt]];
      if (out_d) out_d[obase + sel] = sqrtf(fmaxf(rv[0], 0.0f));
      shead[bt]++;
    }
    __syncthreads();
  }
}

// ---------------------------------------------------------------------------
// Farthest point sampling: one block per batch, dist array in LDS (<=8192 pts).
// ---------------------------------------------------------------------------
__global__ __launch_bounds__(256) void fps_kernel(
    const float* __restrict__ xyz, int N, int npoint, int* __restrict__ out_idx) {
  __shared__ float sdist[8192];
  __shared__ float rv[256];
  __shared__ int   ri[256];
  __shared__ int   cur;
  const int b = blockIdx.x;
  const int tid = threadIdx.x;
  const float* xb = xyz + (size_t)b * N * 3;

  for (int i = tid; i < N; i += 256) sdist[i] = 1e10f;
  if (tid == 0) cur = 0;
  __syncthreads();

  for (int it = 0; it < npoint; ++it) {
    const int far = cur;
    if (tid == 0) out_idx[(size_t)b * npoint + it] = far;
    const float cx = xb[far * 3], cy = xb[far * 3 + 1], cz = xb[far * 3 + 2];
    float bestv = -1.0f; int besti = 0;
    for (int i = tid; i < N; i += 256) {
      float dx = xb[i * 3] - cx, dy = xb[i * 3 + 1] - cy, dz = xb[i * 3 + 2] - cz;
      float nd = fminf(sdist[i], dx * dx + dy * dy + dz * dz);
      sdist[i] = nd;
      if (nd > bestv) { bestv = nd; besti = i; }
    }
    rv[tid] = bestv; ri[tid] = besti;
    __syncthreads();
    for (int s = 128; s > 0; s >>= 1) {
      if (tid < s) {
        if (rv[tid + s] > rv[tid] ||
            (rv[tid + s] == rv[tid] && ri[tid + s] < ri[tid])) {
          rv[tid] = rv[tid + s]; ri[tid] = ri[tid + s];
        }
      }
      __syncthreads();
    }
    if (tid == 0) cur = ri[0];
    __syncthreads();
  }
}

// ---------------------------------------------------------------------------
// LayerNorm (+ optional residual add, optional post-relu). One wave32 per row.
// ---------------------------------------------------------------------------
__global__ __launch_bounds__(256) void ln_kernel(
    const float* __restrict__ x, const float* __restrict__ res,
    const float* __restrict__ g, const float* __restrict__ bta,
    float* __restrict__ out, int M, int D, int postRelu) {
  const int row = blockIdx.x * 8 + (threadIdx.x >> 5);
  if (row >= M) return;
  const int lane = threadIdx.x & 31;
  const int per = D >> 5;              // D in {64,128,256}
  const float* xr = x + (size_t)row * D;
  const float* rr = res ? res + (size_t)row * D : nullptr;

  float vals[8];
  float sum = 0.0f;
  for (int j = 0; j < per; ++j) {
    float v = xr[lane + j * 32];
    if (rr) v += rr[lane + j * 32];
    vals[j] = v; sum += v;
  }
  for (int o = 16; o > 0; o >>= 1) sum += __shfl_xor(sum, o, 32);
  const float mean = sum / (float)D;
  float vs = 0.0f;
  for (int j = 0; j < per; ++j) { float d = vals[j] - mean; vs += d * d; }
  for (int o = 16; o > 0; o >>= 1) vs += __shfl_xor(vs, o, 32);
  const float inv = rsqrtf(vs / (float)D + 1e-5f);
  for (int j = 0; j < per; ++j) {
    int c = lane + j * 32;
    float v = (vals[j] - mean) * inv * g[c] + bta[c];
    if (postRelu) v = fmaxf(v, 0.0f);
    out[(size_t)row * D + c] = v;
  }
}

// ---------------------------------------------------------------------------
// Elementwise / gather kernels. All flat ranges < 2^31 -> 32-bit index math.
// ---------------------------------------------------------------------------
__global__ void slice_cols_kernel(const float* __restrict__ in, float* __restrict__ out,
                                  unsigned total, int Din, int off, int take) {
  unsigned t = blockIdx.x * blockDim.x + threadIdx.x;
  if (t >= total) return;
  unsigned r = t / take; unsigned c = t - r * take;
  out[t] = in[(size_t)r * Din + off + c];
}

__global__ void copy_cols_kernel(const float* __restrict__ in, float* __restrict__ out,
                                 unsigned total, int Din, int Dout, int dstOff) {
  unsigned t = blockIdx.x * blockDim.x + threadIdx.x;
  if (t >= total) return;
  unsigned r = t / Din; unsigned c = t - r * Din;
  out[(size_t)r * Dout + dstOff + c] = in[t];
}

__global__ void gather_rows_kernel(const float* __restrict__ in, const int* __restrict__ idx,
                                   float* __restrict__ out, unsigned total, int rowsPerBatch,
                                   int ninPerBatch, int D, int dstStride, int dstOff) {
  unsigned t = blockIdx.x * blockDim.x + threadIdx.x;
  if (t >= total) return;
  unsigned r = t / D; unsigned c = t - r * D;
  unsigned b = r / rowsPerBatch;
  unsigned src = (unsigned)idx[r];
  out[(size_t)r * dstStride + dstOff + c] = in[((size_t)b * ninPerBatch + src) * D + c];
}

// out[rk, c] = sgn * (xyz_q[row, c] - xyz_ref[idx[rk], c])
__global__ void gather_xyz_diff_kernel(const float* __restrict__ xyz_ref,
                                       const float* __restrict__ xyz_q,
                                       const int* __restrict__ idx, float* __restrict__ out,
                                       unsigned total, int Kn, int nqPerBatch, int nrefPerBatch,
                                       float sgn, int dstStride, int dstOff) {
  unsigned t = blockIdx.x * blockDim.x + threadIdx.x;
  if (t >= total) return;
  unsigned rk = t / 3; unsigned c = t - rk * 3;
  unsigned row = rk / Kn;
  unsigned b = row / nqPerBatch;
  unsigned src = (unsigned)idx[rk];
  float qv = xyz_q[(size_t)row * 3 + c];
  float rv = xyz_ref[((size_t)b * nrefPerBatch + src) * 3 + c];
  out[(size_t)rk * dstStride + dstOff + c] = sgn * (qv - rv);
}

// attn_in[rk, c] = q[row, c] - kk[rk, c] + pe[rk, c]
__global__ void attn_input_kernel(const float* __restrict__ q, const float* __restrict__ kk,
                                  const float* __restrict__ pe, float* __restrict__ out,
                                  unsigned total, int Kn, int D) {
  unsigned t = blockIdx.x * blockDim.x + threadIdx.x;
  if (t >= total) return;
  unsigned rk = t / D; unsigned c = t - rk * D;
  unsigned row = rk / Kn;
  out[t] = q[(size_t)row * D + c] - kk[t] + pe[t];
}

// per-channel softmax over K neighbors + weighted sum of (v + pe)
__global__ void attn_combine_kernel(const float* __restrict__ logits,
                                    const float* __restrict__ v, const float* __restrict__ pe,
                                    float* __restrict__ out, unsigned total, int Kn, int D) {
  unsigned t = blockIdx.x * blockDim.x + threadIdx.x;
  if (t >= total) return;
  unsigned row = t / D; unsigned c = t - row * D;
  const float* l  = logits + (size_t)row * Kn * D + c;
  const float* vv = v      + (size_t)row * Kn * D + c;
  const float* pp = pe     + (size_t)row * Kn * D + c;
  float mx = -3.4e38f;
  for (int k = 0; k < Kn; ++k) mx = fmaxf(mx, l[(size_t)k * D]);
  float s = 0.0f;
  for (int k = 0; k < Kn; ++k) s += __expf(l[(size_t)k * D] - mx);
  const float invs = 1.0f / s;
  float o = 0.0f;
  for (int k = 0; k < Kn; ++k) {
    float w = __expf(l[(size_t)k * D] - mx) * invs;
    o += w * (vv[(size_t)k * D] + pp[(size_t)k * D]);
  }
  out[t] = o;
}

__global__ void maxpool_kernel(const float* __restrict__ in, float* __restrict__ out,
                               unsigned total, int Kn, int D) {
  unsigned t = blockIdx.x * blockDim.x + threadIdx.x;
  if (t >= total) return;
  unsigned row = t / D; unsigned c = t - row * D;
  float m = -3.4e38f;
  for (int k = 0; k < Kn; ++k) m = fmaxf(m, in[((size_t)row * Kn + k) * D + c]);
  out[t] = m;
}

// inverse-distance 3-NN interpolation
__global__ void interp_kernel(const float* __restrict__ feat_lo, const float* __restrict__ dists,
                              const int* __restrict__ idx, float* __restrict__ out,
                              unsigned total, int nqPerBatch, int nloPerBatch, int D) {
  unsigned t = blockIdx.x * blockDim.x + threadIdx.x;
  if (t >= total) return;
  unsigned row = t / D; unsigned c = t - row * D;
  unsigned b = row / nqPerBatch;
  const float* dd = dists + (size_t)row * 3;
  const int*   ii = idx   + (size_t)row * 3;
  float w0 = 1.0f / (dd[0] + 1e-8f);
  float w1 = 1.0f / (dd[1] + 1e-8f);
  float w2 = 1.0f / (dd[2] + 1e-8f);
  float inv = 1.0f / (w0 + w1 + w2);
  const float* fb = feat_lo + (size_t)b * nloPerBatch * D;
  out[t] = (w0 * fb[(size_t)ii[0] * D + c] +
            w1 * fb[(size_t)ii[1] * D + c] +
            w2 * fb[(size_t)ii[2] * D + c]) * inv;
}

// ---------------------------------------------------------------------------
// Host-side orchestration
// ---------------------------------------------------------------------------
struct MLP2 { const float *w1, *b1, *w2, *b2; };
struct PTP  { const float *wq, *wk, *wv; MLP2 pos, att; const float *g, *b; };

static inline unsigned ceilu(long a, long b) { return (unsigned)((a + b - 1) / b); }

static void gemm(const float* A, const float* W, const float* bias, float* C,
                 long M, int N, int K, int relu, hipStream_t s) {
  dim3 grid(ceilu(M, GBM), (unsigned)(N / GBN));     // all N are multiples of 64
  if (relu) gemm_wmma_kernel<1><<<grid, 128, 0, s>>>(A, W, bias, C, (int)M, N, K);
  else      gemm_wmma_kernel<0><<<grid, 128, 0, s>>>(A, W, bias, C, (int)M, N, K);
}

extern "C" void kernel_launch(void* const* d_in, const int* in_sizes, int n_in,
                              void* d_out, int out_size, void* d_ws, size_t ws_size,
                              hipStream_t stream) {
  (void)in_sizes; (void)n_in; (void)out_size; (void)ws_size;
  const int Bc = 4, Nc = 8192, DIN = 6, D0 = 64, D1 = 128, D2 = 256;
  const int NP1 = 512, NP2 = 128, KA = 16, KD = 16, DOUT = 256;

  // ---- parameter leaves, in setup_inputs() recursive dict (insertion) order
  const float* x = (const float*)d_in[0];
  int ii = 1;
  auto nf  = [&]() -> const float* { return (const float*)d_in[ii++]; };
  auto ldM = [&]() { MLP2 m; m.w1 = nf(); m.b1 = nf(); m.w2 = nf(); m.b2 = nf(); return m; };
  auto ldP = [&]() { PTP p; p.wq = nf(); p.wk = nf(); p.wv = nf();
                     p.pos = ldM(); p.att = ldM(); p.g = nf(); p.b = nf(); return p; };

  const float *stem_w = nf(), *stem_b = nf(), *stem_g = nf(), *stem_lb = nf();
  PTP pt0 = ldP(), pt1 = ldP(), pt2 = ldP();
  MLP2 td1_mlp = ldM(); const float *td1_g = nf(), *td1_b = nf();
  MLP2 td2_mlp = ldM(); const float *td2_g = nf(), *td2_b = nf();
  MLP2 tu1_fc  = ldM(); const float *tu1_g = nf(), *tu1_b = nf();
  PTP ptd1 = ldP();
  MLP2 tu2_fc  = ldM(); const float *tu2_g = nf(), *tu2_b = nf();
  PTP ptd2 = ldP();
  const float *head_w = nf(), *head_b = nf(), *head_g = nf(), *head_lb = nf();

  // ---- workspace bump allocator (ping-pong BIG buffers reused per stage)
  char* base = (char*)d_ws;
  size_t off = 0;
  auto allocF = [&](size_t n) { float* p = (float*)(base + off);
                                off += ((n * 4 + 255) & ~(size_t)255); return p; };
  auto allocI = [&](size_t n) { int* p = (int*)(base + off);
                                off += ((n * 4 + 255) & ~(size_t)255); return p; };
  const size_t BIGCAP = (size_t)Bc * Nc * KA * D0;      // 33.5M floats
  float *BIG1 = allocF(BIGCAP), *BIG2 = allocF(BIGCAP);
  float *BIG3 = allocF(BIGCAP), *BIG4 = allocF(BIGCAP);
  float *xyz0 = allocF((size_t)Bc * Nc * 3);
  float *xyz1 = allocF((size_t)Bc * NP1 * 3);
  float *xyz2 = allocF((size_t)Bc * NP2 * 3);
  float *f0   = allocF((size_t)Bc * Nc * D0);
  float *f0u  = allocF((size_t)Bc * Nc * D0);
  float *f1   = allocF((size_t)Bc * NP1 * D1);
  float *f1u  = allocF((size_t)Bc * NP1 * D1);
  float *f2   = allocF((size_t)Bc * NP2 * D2);
  float *hpre = allocF((size_t)Bc * Nc * DOUT);
  float *qbuf = allocF((size_t)Bc * Nc * D1);           // q / combine / interp scratch
  float *pein = allocF((size_t)Bc * Nc * KA * 3);
  float *dist3 = allocF((size_t)Bc * Nc * 3);
  int *idxK = allocI((size_t)Bc * Nc * KA);
  int *idx3 = allocI((size_t)Bc * Nc * 3);
  int *fpsi = allocI((size_t)Bc * NP1);

  auto ln = [&](const float* xin, const float* res, const float* g, const float* bta,
                float* o, long M, int D, int relu) {
    ln_kernel<<<ceilu(M, 8), 256, 0, stream>>>(xin, res, g, bta, o, (int)M, D, relu);
  };

  // ---- point-transformer layer (in-place update of feat)
  auto run_pt = [&](const PTP& p, const float* xyz, float* feat, int Np, int D) {
    long rows = (long)Bc * Np;
    long RK = rows * KA;
    knn_kernel<16><<<dim3((unsigned)Np, Bc), 128, 0, stream>>>(
        xyz, xyz, Np, Np, idxK, (float*)nullptr);
    gather_xyz_diff_kernel<<<ceilu(RK * 3, 256), 256, 0, stream>>>(
        xyz, xyz, idxK, pein, (unsigned)(RK * 3), KA, Np, Np, 1.0f, 3, 0);
    gather_rows_kernel<<<ceilu(RK * D, 256), 256, 0, stream>>>(
        feat, idxK, BIG1, (unsigned)(RK * D), Np * KA, Np, D, D, 0);  // knn_feat
    gemm(BIG1, p.wk, nullptr, BIG2, RK, D, D, 0, stream);             // k
    gemm(BIG1, p.wv, nullptr, BIG3, RK, D, D, 0, stream);             // v
    gemm(pein, p.pos.w1, p.pos.b1, BIG1, RK, D, 3, 1, stream);        // pe hidden
    gemm(BIG1, p.pos.w2, p.pos.b2, BIG4, RK, D, D, 0, stream);        // pe
    gemm(feat, p.wq, nullptr, qbuf, rows, D, D, 0, stream);           // q
    attn_input_kernel<<<ceilu(RK * D, 256), 256, 0, stream>>>(
        qbuf, BIG2, BIG4, BIG1, (unsigned)(RK * D), KA, D);           // q-k+pe
    gemm(BIG1, p.att.w1, p.att.b1, BIG2, RK, D, D, 1, stream);        // att hidden
    gemm(BIG2, p.att.w2, p.att.b2, BIG1, RK, D, D, 0, stream);        // logits
    attn_combine_kernel<<<ceilu(rows * D, 256), 256, 0, stream>>>(
        BIG1, BIG3, BIG4, qbuf, (unsigned)(rows * D), KA, D);
    ln(qbuf, feat, p.g, p.b, feat, rows, D, 0);                       // LN(out+feat)
  };

  // ---- transition down
  auto run_td = [&](const MLP2& m, const float* g, const float* bta,
                    const float* xyz, const float* feat, int Np, int npoint,
                    int D, int Dh, float* new_xyz, float* new_feat) {
    long rows = (long)Bc * npoint;
    long RK = rows * KD;
    int Dg = 3 + D;
    fps_kernel<<<Bc, 256, 0, stream>>>(xyz, Np, npoint, fpsi);
    gather_rows_kernel<<<ceilu(rows * 3, 256), 256, 0, stream>>>(
        xyz, fpsi, new_xyz, (unsigned)(rows * 3), npoint, Np, 3, 3, 0);
    knn_kernel<16><<<dim3((unsigned)npoint, Bc), 128, 0, stream>>>(
        xyz, new_xyz, Np, npoint, idxK, (float*)nullptr);
    gather_xyz_diff_kernel<<<ceilu(RK * 3, 256), 256, 0, stream>>>(
        xyz, new_xyz, idxK, BIG1, (unsigned)(RK * 3), KD, npoint, Np, -1.0f, Dg, 0);
    gather_rows_kernel<<<ceilu(RK * D, 256), 256, 0, stream>>>(
        feat, idxK, BIG1, (unsigned)(RK * D), npoint * KD, Np, D, Dg, 3);
    gemm(BIG1, m.w1, m.b1, BIG2, RK, Dh, Dg, 1, stream);
    gemm(BIG2, m.w2, m.b2, BIG3, RK, Dh, Dh, 0, stream);
    maxpool_kernel<<<ceilu(rows * Dh, 256), 256, 0, stream>>>(
        BIG3, new_feat, (unsigned)(rows * Dh), KD, Dh);
    ln(new_feat, nullptr, g, bta, new_feat, rows, Dh, 0);
  };

  // ---- transition up
  auto run_tu = [&](const MLP2& m, const float* g, const float* bta,
                    const float* xyz_hi, const float* xyz_lo,
                    const float* feat_skip, const float* feat_lo,
                    int Nhi, int Nlo, int Dskip, int Dlo, int Do, float* out) {
    long rows = (long)Bc * Nhi;
    int Dfused = Dskip + Dlo;
    knn_kernel<3><<<dim3((unsigned)Nhi, Bc), 128, 0, stream>>>(
        xyz_lo, xyz_hi, Nlo, Nhi, idx3, dist3);
    interp_kernel<<<ceilu(rows * Dlo, 256), 256, 0, stream>>>(
        feat_lo, dist3, idx3, qbuf, (unsigned)(rows * Dlo), Nhi, Nlo, Dlo);
    copy_cols_kernel<<<ceilu(rows * Dskip, 256), 256, 0, stream>>>(
        feat_skip, BIG1, (unsigned)(rows * Dskip), Dskip, Dfused, 0);
    copy_cols_kernel<<<ceilu(rows * Dlo, 256), 256, 0, stream>>>(
        qbuf, BIG1, (unsigned)(rows * Dlo), Dlo, Dfused, Dskip);
    gemm(BIG1, m.w1, m.b1, BIG2, rows, Do, Dfused, 1, stream);
    gemm(BIG2, m.w2, m.b2, out, rows, Do, Do, 0, stream);
    ln(out, nullptr, g, bta, out, rows, Do, 0);
  };

  // ---- forward
  const long rows0 = (long)Bc * Nc;
  slice_cols_kernel<<<ceilu(rows0 * 3, 256), 256, 0, stream>>>(
      x, xyz0, (unsigned)(rows0 * 3), DIN, 0, 3);
  gemm(x, stem_w, stem_b, f0, rows0, D0, DIN, 0, stream);
  ln(f0, nullptr, stem_g, stem_lb, f0, rows0, D0, 1);                // relu(ln(.))

  run_pt(pt0, xyz0, f0, Nc, D0);
  run_td(td1_mlp, td1_g, td1_b, xyz0, f0, Nc, NP1, D0, D1, xyz1, f1);
  run_pt(pt1, xyz1, f1, NP1, D1);
  run_td(td2_mlp, td2_g, td2_b, xyz1, f1, NP1, NP2, D1, D2, xyz2, f2);
  run_pt(pt2, xyz2, f2, NP2, D2);
  run_tu(tu1_fc, tu1_g, tu1_b, xyz1, xyz2, f1, f2, NP1, NP2, D1, D2, D1, f1u);
  run_pt(ptd1, xyz1, f1u, NP1, D1);
  run_tu(tu2_fc, tu2_g, tu2_b, xyz0, xyz1, f0, f1u, Nc, NP1, D0, D1, D0, f0u);
  run_pt(ptd2, xyz0, f0u, Nc, D0);

  gemm(f0u, head_w, head_b, hpre, rows0, DOUT, D0, 0, stream);
  ln(hpre, nullptr, head_g, head_lb, (float*)d_out, rows0, DOUT, 0);
}
